// CustomMSDeformableAttention_2001454760356
// MI455X (gfx1250) — compile-verified, matrix operand order checked by reading, not run on hardware
//
#include <hip/hip_runtime.h>
#include <hip/hip_bf16.h>

// ---------------------------------------------------------------------------
// MS-Deformable Attention for MI455X (gfx1250).
//  - value projection: bf16x3 split-precision GEMM on v_wmma_f32_16x16x32_bf16
//    (fp32-class accuracy, 32-deep MACs per instruction)
//  - offset / attention / output GEMMs: true fp32 v_wmma_f32_16x16x4_f32
//  - A-tiles staged with GLOBAL_LOAD_ASYNC_TO_LDS_B128 (ASYNCcnt-tracked)
// ---------------------------------------------------------------------------

#define EMBED     256
#define HEADS     8
#define LEVELS    4
#define POINTS    4
#define HEAD_DIM  32        // EMBED / HEADS
#define BS        8
#define NQ        900
#define NV        13294     // 100*100 + 50*50 + 25*25 + 13*13
#define NBQ       (BS * NQ)             // 7200  (divisible by 16)
#define MVAL      (BS * NV)             // 106352 (divisible by 16)
#define NPT       (BS * NQ * HEADS * LEVELS * POINTS)  // 921600

#define LDS_STRIDE 260      // 260 % 64 == 4 -> conflict-free fragment reads

typedef __attribute__((ext_vector_type(2)))  float  v2f;
typedef __attribute__((ext_vector_type(8)))  float  v8f;
typedef __attribute__((ext_vector_type(16))) __bf16 v16bf;
typedef __attribute__((ext_vector_type(4)))  int    v4i;

#define AS_GLOBAL __attribute__((address_space(1)))
#define AS_LDS    __attribute__((address_space(3)))

// ---------------------------------------------------------------------------
// Stage a 16 x 256 fp32 A-tile into LDS (stride LDS_STRIDE).
// 256 threads, 64 bytes each, moved with GLOBAL_LOAD_ASYNC_TO_LDS_B128
// (direct global->LDS, no VGPR staging), completed with s_wait_asynccnt.
// ---------------------------------------------------------------------------
__device__ __forceinline__ void stage_A_tile(const float* __restrict__ A,
                                             int m0, float* As)
{
    const int row  = threadIdx.x >> 4;          // 0..15
    const int cseg = (threadIdx.x & 15) * 16;   // 0,16,...,240
    const float* src = A + (size_t)(m0 + row) * EMBED + cseg;
    float*       dst = As + row * LDS_STRIDE + cseg;

#if __has_builtin(__builtin_amdgcn_global_load_async_to_lds_b128)
#pragma unroll
    for (int i = 0; i < 4; ++i) {
        __builtin_amdgcn_global_load_async_to_lds_b128(
            (AS_GLOBAL v4i*)(src + 4 * i),
            (AS_LDS    v4i*)(dst + 4 * i),
            /*offset=*/0, /*cpol=*/0);
    }
#if __has_builtin(__builtin_amdgcn_s_wait_asynccnt)
    __builtin_amdgcn_s_wait_asynccnt(0);
#else
    asm volatile("s_wait_asynccnt 0x0" ::: "memory");
#endif
#else
    const float4* s4 = (const float4*)src;
    float4*       d4 = (float4*)dst;
    d4[0] = s4[0]; d4[1] = s4[1]; d4[2] = s4[2]; d4[3] = s4[3];
#endif
    __syncthreads();
}

// ---------------------------------------------------------------------------
// fp32 WMMA GEMM, K = 256: out[m*N+n] = A@B + bias[n] (+ residual[m*N+n]).
// 8 waves/block, one 16x16 tile per wave; grid.x covers N in 128-col chunks,
// grid.y = M/16. Used for the precision-sensitive small GEMMs.
// ---------------------------------------------------------------------------
__global__ __launch_bounds__(256)
void wmma_gemm_f32_k256(const float* __restrict__ A,
                        const float* __restrict__ B,
                        const float* __restrict__ bias,
                        const float* __restrict__ residual,
                        float* __restrict__ out,
                        int N)
{
    __shared__ float As[16 * LDS_STRIDE];
    const int m0 = blockIdx.y * 16;
    stage_A_tile(A, m0, As);

    const int wave = threadIdx.x >> 5;
    const int lane = threadIdx.x & 31;
    const int l15  = lane & 15;
    const int half = lane >> 4;                 // K-pair {0,1} vs {2,3}
    const int n0   = (blockIdx.x * 8 + wave) * 16;
    const int n    = n0 + l15;

    const float bb = bias ? bias[n] : 0.0f;
    v8f c = {bb, bb, bb, bb, bb, bb, bb, bb};

    const float* as_row = As + (size_t)l15 * LDS_STRIDE + 2 * half;

#pragma unroll 8
    for (int k0 = 0; k0 < EMBED; k0 += 4) {
        v2f a;                                   // A 16x4: lane = M row
        a.x = as_row[k0];
        a.y = as_row[k0 + 1];
        const int kb = k0 + 2 * half;            // B 4x16: lane = N col
        v2f b;
        b.x = B[(size_t)kb * N + n];
        b.y = B[(size_t)(kb + 1) * N + n];
        c = __builtin_amdgcn_wmma_f32_16x16x4_f32(
                false, a, false, b, (short)0, c, false, false);
    }

#pragma unroll
    for (int r = 0; r < 8; ++r) {
        const int m = m0 + r + 8 * half;
        float val = c[r];
        if (residual) val += residual[(size_t)m * N + n];
        out[(size_t)m * N + n] = val;
    }
}

// ---------------------------------------------------------------------------
// Value projection, bf16x3 split precision on v_wmma_f32_16x16x32_bf16.
//   a = ah + al (ah = bf16 RNE, al = bf16(a - ah)); a.b ~= ah.bh + ah.bl + al.bh
// Residual term al*bl (~2^-17 relative) is dropped -> fp32-class accuracy.
// Output scattered head-major: out[((b*H+h)*NV + i)*32 + d].
// A fragment (16-bit, 16x32): lane = M row; half=0 -> K {0..7,16..23},
// half=1 -> K {8..15,24..31}. B mirrors with lane = N col.
// ---------------------------------------------------------------------------
__global__ __launch_bounds__(256)
void wmma_value_proj_bf16x3(const float* __restrict__ A,
                            const float* __restrict__ B,
                            const float* __restrict__ bias,
                            float* __restrict__ out)
{
    __shared__ float As[16 * LDS_STRIDE];
    const int m0 = blockIdx.y * 16;
    stage_A_tile(A, m0, As);

    const int wave = threadIdx.x >> 5;
    const int lane = threadIdx.x & 31;
    const int l15  = lane & 15;
    const int half = lane >> 4;
    const int n0   = (blockIdx.x * 8 + wave) * 16;
    const int n    = n0 + l15;
    const int N    = EMBED;

    const float bb = bias[n];
    v8f c = {bb, bb, bb, bb, bb, bb, bb, bb};

    const float* ap = As + (size_t)l15 * LDS_STRIDE + 8 * half;

    for (int k0 = 0; k0 < EMBED; k0 += 32) {
        // ---- A fragment: two runs of 8 consecutive K from LDS ------------
        float af[16];
#pragma unroll
        for (int e = 0; e < 8; ++e) {
            af[e]     = ap[k0 + e];
            af[8 + e] = ap[k0 + 16 + e];
        }
        v16bf ah, al;
#pragma unroll
        for (int e = 0; e < 16; ++e) {
            const __bf16 h = (__bf16)af[e];
            ah[e] = h;
            al[e] = (__bf16)(af[e] - (float)h);
        }
        // ---- B fragment: 16 rows of B at column n ------------------------
        const int kb = k0 + 8 * half;
        v16bf bh, bl;
#pragma unroll
        for (int e = 0; e < 8; ++e) {
            const float b0 = B[(size_t)(kb + e) * N + n];
            const float b1 = B[(size_t)(kb + 16 + e) * N + n];
            const __bf16 h0 = (__bf16)b0;
            const __bf16 h1 = (__bf16)b1;
            bh[e]     = h0;
            bh[8 + e] = h1;
            bl[e]     = (__bf16)(b0 - (float)h0);
            bl[8 + e] = (__bf16)(b1 - (float)h1);
        }
        // ---- three bf16 WMMAs accumulate the split product ---------------
        c = __builtin_amdgcn_wmma_f32_16x16x32_bf16(
                false, ah, false, bh, (short)0, c, false, false);
        c = __builtin_amdgcn_wmma_f32_16x16x32_bf16(
                false, ah, false, bl, (short)0, c, false, false);
        c = __builtin_amdgcn_wmma_f32_16x16x32_bf16(
                false, al, false, bh, (short)0, c, false, false);
    }

#pragma unroll
    for (int r = 0; r < 8; ++r) {
        const int m  = m0 + r + 8 * half;
        const int b_ = m / NV;
        const int i_ = m - b_ * NV;
        const int h_ = n >> 5;
        const int d_ = n & 31;
        out[(((size_t)b_ * HEADS + h_) * NV + i_) * HEAD_DIM + d_] = c[r];
    }
}

// ---------------------------------------------------------------------------
// Per-(b,q,h) thread: softmax over 16 logits + sampling locations.
// ---------------------------------------------------------------------------
__global__ __launch_bounds__(256)
void msda_softmax_loc(const float* __restrict__ aw_raw,
                      const float* __restrict__ off,
                      const float* __restrict__ refpts,
                      const int*   __restrict__ sshapes,
                      float* __restrict__ locx,
                      float* __restrict__ locy,
                      float* __restrict__ wts)
{
    const int t = blockIdx.x * blockDim.x + threadIdx.x;   // (b*NQ+q)*HEADS + h
    if (t >= BS * NQ * HEADS) return;
    const int h  = t % HEADS;
    const int bq = t / HEADS;

    const float* logits = aw_raw + (size_t)bq * (HEADS * LEVELS * POINTS) + h * 16;
    float w[16];
    float mx = logits[0];
#pragma unroll
    for (int j = 1; j < 16; ++j) mx = fmaxf(mx, logits[j]);
    float sum = 0.0f;
#pragma unroll
    for (int j = 0; j < 16; ++j) { w[j] = __expf(logits[j] - mx); sum += w[j]; }
    const float inv = 1.0f / sum;

    const float* o  = off + (size_t)bq * EMBED + h * (LEVELS * POINTS * 2);
    const float* rp = refpts + (size_t)bq * (LEVELS * 2);

#pragma unroll
    for (int l = 0; l < LEVELS; ++l) {
        const float nw = (float)sshapes[2 * l + 1];   // W normalizes x
        const float nh = (float)sshapes[2 * l];       // H normalizes y
        const float rx = rp[2 * l];
        const float ry = rp[2 * l + 1];
#pragma unroll
        for (int p = 0; p < POINTS; ++p) {
            const int j   = l * POINTS + p;
            const int idx = t * (LEVELS * POINTS) + j;
            locx[idx] = rx + o[2 * j]     / nw;
            locy[idx] = ry + o[2 * j + 1] / nh;
            wts[idx]  = w[j] * inv;
        }
    }
}

// ---------------------------------------------------------------------------
// Bilinear gather. One wave per (b,q,h); lane = channel d (HEAD_DIM == 32).
// Head-major vproj -> each corner is one coalesced 128B row.
// ---------------------------------------------------------------------------
__global__ __launch_bounds__(256)
void msda_sample(const float* __restrict__ vproj,
                 const float* __restrict__ locx,
                 const float* __restrict__ locy,
                 const float* __restrict__ wts,
                 const int*   __restrict__ sshapes,
                 float* __restrict__ out_s)
{
    const int wid  = (blockIdx.x * blockDim.x + threadIdx.x) >> 5;
    const int lane = threadIdx.x & 31;
    if (wid >= BS * NQ * HEADS) return;

    const int h  = wid % HEADS;
    const int bq = wid / HEADS;
    const int b  = bq / NQ;

    const float* vh   = vproj + ((size_t)b * HEADS + h) * NV * HEAD_DIM;
    const int    base = wid * (LEVELS * POINTS);

    float acc = 0.0f;
    int start = 0;
#pragma unroll
    for (int l = 0; l < LEVELS; ++l) {
        const int H = sshapes[2 * l];
        const int W = sshapes[2 * l + 1];
#pragma unroll
        for (int p = 0; p < POINTS; ++p) {
            const int   j  = base + l * POINTS + p;
            const float aw = wts[j];
            const float x  = locx[j] * (float)W - 0.5f;
            const float y  = locy[j] * (float)H - 0.5f;
            const float x0 = floorf(x);
            const float y0 = floorf(y);
            const float lx = x - x0;
            const float ly = y - y0;
            const int   xi = (int)x0;
            const int   yi = (int)y0;
#pragma unroll
            for (int cc = 0; cc < 4; ++cc) {
                const int dx = cc & 1;
                const int dy = cc >> 1;
                const int xx = xi + dx;
                const int yy = yi + dy;
                const float cw = (dx ? lx : 1.0f - lx) * (dy ? ly : 1.0f - ly) * aw;
                const bool valid = (xx >= 0) & (xx < W) & (yy >= 0) & (yy < H);
                const int xc = min(max(xx, 0), W - 1);
                const int yc = min(max(yy, 0), H - 1);
                const float vv = vh[((size_t)(start + yc * W + xc)) * HEAD_DIM + lane];
                acc += valid ? cw * vv : 0.0f;
            }
        }
        start += H * W;
    }
    out_s[(size_t)bq * EMBED + h * HEAD_DIM + lane] = acc;
}

// ---------------------------------------------------------------------------
// Host-side orchestration (graph-capture safe: only kernel launches).
// ---------------------------------------------------------------------------
extern "C" void kernel_launch(void* const* d_in, const int* in_sizes, int n_in,
                              void* d_out, int out_size, void* d_ws, size_t ws_size,
                              hipStream_t stream)
{
    const float* query   = (const float*)d_in[0];   // (BS, NQ, 256)
    const float* value   = (const float*)d_in[1];   // (BS, NV, 256)
    const float* refpts  = (const float*)d_in[2];   // (BS, NQ, LEVELS, 2)
    const int*   sshapes = (const int*)  d_in[3];   // (LEVELS, 2)
    const float* W_off   = (const float*)d_in[4];   // (256, 256)
    const float* b_off   = (const float*)d_in[5];   // (256,)
    const float* W_attn  = (const float*)d_in[6];   // (256, 128)
    const float* b_attn  = (const float*)d_in[7];   // (128,)
    const float* W_v     = (const float*)d_in[8];   // (256, 256)
    const float* b_v     = (const float*)d_in[9];   // (256,)
    const float* W_out   = (const float*)d_in[10];  // (256, 256)
    const float* b_out   = (const float*)d_in[11];  // (256,)
    float*       out     = (float*)d_out;           // (BS, NQ, 256)

    // ---- workspace carve-out (floats) -----------------------------------
    float* ws      = (float*)d_ws;
    float* vproj   = ws;                                       // BS*HEADS*NV*32
    float* off_raw = vproj   + (size_t)BS * HEADS * NV * HEAD_DIM;
    float* aw_raw  = off_raw + (size_t)NBQ * EMBED;            // NBQ*128
    float* locx    = aw_raw  + (size_t)NBQ * (HEADS * LEVELS * POINTS);
    float* locy    = locx    + (size_t)NPT;
    float* wts     = locy    + (size_t)NPT;
    float* out_s   = wts     + (size_t)NPT;                    // NBQ*EMBED

    // 1. value projection (bf16x3 split precision) -> head-major (b,h,NV,32)
    wmma_value_proj_bf16x3<<<dim3(EMBED / 128, MVAL / 16), 256, 0, stream>>>(
        value, W_v, b_v, vproj);

    // 2. sampling-offset projection (fp32 WMMA, N = 256)
    wmma_gemm_f32_k256<<<dim3(EMBED / 128, NBQ / 16), 256, 0, stream>>>(
        query, W_off, b_off, nullptr, off_raw, EMBED);

    // 3. attention-logit projection (fp32 WMMA, N = 128)
    wmma_gemm_f32_k256<<<dim3(1, NBQ / 16), 256, 0, stream>>>(
        query, W_attn, b_attn, nullptr, aw_raw, HEADS * LEVELS * POINTS);

    // 4. softmax + sampling locations
    msda_softmax_loc<<<(BS * NQ * HEADS + 255) / 256, 256, 0, stream>>>(
        aw_raw, off_raw, refpts, sshapes, locx, locy, wts);

    // 5. bilinear gather (one wave per (b,q,h))
    msda_sample<<<(BS * NQ * HEADS) / 8, 256, 0, stream>>>(
        vproj, locx, locy, wts, sshapes, out_s);

    // 6. output projection + residual (fp32 WMMA)
    wmma_gemm_f32_k256<<<dim3(EMBED / 128, NBQ / 16), 256, 0, stream>>>(
        out_s, W_out, b_out, /*residual=*/query, out, EMBED);
}
